// LanguageModel_7773890806382
// MI455X (gfx1250) — matrix operand order repaired and checked by reading, not compiled, not run on hardware
//
#include <hip/hip_runtime.h>
#include <hip/hip_bf16.h>
#include <math.h>

// ---------------------------------------------------------------------------
// GPT forward pass for MI455X (gfx1250), wave32 + WMMA bf16 (f32 accumulate).
// Model: V=32000, D=1024, H=16, L=8, B=4, T=1024, HD=64.
// Compute-bound (~1.23 TFLOP vs ~23us of HBM traffic at 23.3 TB/s), so all
// matmuls route through v_wmma_f32_16x16x32_bf16 with in-register fp32->bf16
// conversion of weights/activations.
// ---------------------------------------------------------------------------

constexpr int kV  = 32000;
constexpr int kD  = 1024;
constexpr int kH  = 16;
constexpr int kL  = 8;
constexpr int kT  = 1024;
constexpr int kB  = 4;
constexpr int kHD = 64;
constexpr int kM  = kB * kT;        // 4096 rows of activations
constexpr float kEPS = 1e-5f;

typedef __attribute__((ext_vector_type(16))) __bf16 v16bf;
typedef __attribute__((ext_vector_type(8)))  float  v8f;

__device__ __forceinline__ __bf16 tobf(float f) { return (__bf16)f; }

__device__ __forceinline__ void cvt4(v16bf& v, int base, float4 f) {
    v[base + 0] = tobf(f.x);
    v[base + 1] = tobf(f.y);
    v[base + 2] = tobf(f.z);
    v[base + 3] = tobf(f.w);
}

// ---------------------------------------------------------------------------
// GEMM: C[M,N] = act(A[M,K] @ W[N,K]^T + bias) (+ residual)
// A fp32 row-major [M,K], W fp32 row-major [N,K] (torch Linear layout).
// Block = 256 threads (8 waves). Block tile 128(M) x 128(N).
// Wave (wr,wc): wr in 0..3 -> 32 rows (2 A-fragments), wc in 0..1 -> 64 cols
// (4 WMMA col-tiles). 8 WMMAs per 32-K step, each B-fragment feeds 2 WMMAs.
// ---------------------------------------------------------------------------
__global__ __launch_bounds__(256) void gemm_bf16w(
    const float* __restrict__ A, const float* __restrict__ W,
    const float* __restrict__ bias, const float* __restrict__ residual,
    float* __restrict__ C, int N, int K, int relu)
{
    const int lane    = threadIdx.x & 31;
    const int wave    = threadIdx.x >> 5;
    const int wr      = wave >> 1;
    const int wc      = wave & 1;
    const int m0      = blockIdx.y * 128 + wr * 32;
    const int n0      = blockIdx.x * 128 + wc * 64;
    const int arow    = lane & 15;
    const int halfsel = lane >> 4;          // 0: lanes 0-15, 1: lanes 16-31

    v8f acc[8] = {};
    const float* Arow0 = A + (size_t)(m0 + arow) * K;
    const float* Arow1 = A + (size_t)(m0 + 16 + arow) * K;

    for (int k = 0; k < K; k += 32) {
        // ---- A fragments (16x32 bf16 each): lane holds row (lane&15),
        //      K = kb+0..15 contiguous with kb = k + halfsel*16
        const int kb = k + halfsel * 16;
        v16bf af0, af1;
        {
            const float4* p0 = reinterpret_cast<const float4*>(Arow0 + kb);
            const float4* p1 = reinterpret_cast<const float4*>(Arow0 + kb + 8);
            cvt4(af0, 0,  p0[0]);
            cvt4(af0, 4,  p0[1]);
            cvt4(af0, 8,  p1[0]);
            cvt4(af0, 12, p1[1]);
            const float4* q0 = reinterpret_cast<const float4*>(Arow1 + kb);
            const float4* q1 = reinterpret_cast<const float4*>(Arow1 + kb + 8);
            cvt4(af1, 0,  q0[0]);
            cvt4(af1, 4,  q0[1]);
            cvt4(af1, 8,  q1[0]);
            cvt4(af1, 12, q1[1]);
        }

        // ---- B fragments (32x16 bf16): lane holds col (lane&15),
        //      K = kbase+0..15 contiguous, kbase = k + halfsel*16
        const int kbase = k + halfsel * 16;
#pragma unroll
        for (int t = 0; t < 4; ++t) {
            const float* Wrow = W + (size_t)(n0 + t * 16 + arow) * K + kbase;
            // Speculative prefetch of the next K-slice of this weight row
            // (global_prefetch_b8; dropped silently if past the tensor end).
            __builtin_prefetch(Wrow + 32, 0, 1);
            const float4* wp = reinterpret_cast<const float4*>(Wrow);
            v16bf bf;
            cvt4(bf, 0,  wp[0]);
            cvt4(bf, 4,  wp[1]);
            cvt4(bf, 8,  wp[2]);
            cvt4(bf, 12, wp[3]);
            acc[t] = __builtin_amdgcn_wmma_f32_16x16x32_bf16(
                false, af0, false, bf, (short)0, acc[t], false, false);
            acc[4 + t] = __builtin_amdgcn_wmma_f32_16x16x32_bf16(
                false, af1, false, bf, (short)0, acc[4 + t], false, false);
        }
    }

    // ---- Epilogue. C layout: VGPR e -> M = e + halfsel*8, N = lane&15.
#pragma unroll
    for (int g = 0; g < 2; ++g) {
#pragma unroll
        for (int t = 0; t < 4; ++t) {
#pragma unroll
            for (int e = 0; e < 8; ++e) {
                const int row = m0 + g * 16 + e + halfsel * 8;
                const int col = n0 + t * 16 + arow;
                float val = acc[g * 4 + t][e];
                if (bias)  val += bias[col];
                if (relu)  val = fmaxf(val, 0.0f);
                const size_t idx = (size_t)row * N + col;
                if (residual) val += residual[idx];
                C[idx] = val;
            }
        }
    }
}

// ---------------------------------------------------------------------------
// Flash attention (causal). One wave per (b, h, 16-query tile).
// kqv layout: [B, T, 3D] with split order k | q | v, head h at offset h*64.
// Online softmax over 32-key blocks; P transposed through LDS into the
// A-fragment layout; V staged transposed as bf16 in LDS for B-fragments.
// ---------------------------------------------------------------------------
__global__ __launch_bounds__(32) void attention_kernel(
    const float* __restrict__ kqv, float* __restrict__ out)
{
    __shared__ __bf16 vT[64 * 32];      // [d][key] transposed V tile (32 keys)
    __shared__ __bf16 pbuf[16 * 32];    // [m][key] probability tile

    const int lane    = threadIdx.x;
    const int qt      = blockIdx.x;     // 0..63 query tile
    const int bh      = blockIdx.y;     // 0..63
    const int b       = bh / kH;
    const int h       = bh % kH;
    const int q0      = qt * 16;
    const int arow    = lane & 15;
    const int halfsel = lane >> 4;

    const size_t rstride = (size_t)3 * kD;
    const float* base = kqv + (size_t)b * kT * rstride;

    // ---- Q fragments for K-dim 0..31 and 32..63, pre-scaled by HD^-0.5
    v16bf qa0, qa1;
    {
        const float* qrow = base + (size_t)(q0 + arow) * rstride + kD + h * kHD;
        const int kb = halfsel * 16;
#pragma unroll
        for (int i = 0; i < 8; ++i) {
            qa0[i]     = tobf(qrow[kb + i]       * 0.125f);
            qa0[8 + i] = tobf(qrow[kb + 8 + i]  * 0.125f);
            qa1[i]     = tobf(qrow[32 + kb + i]      * 0.125f);
            qa1[8 + i] = tobf(qrow[32 + kb + 8 + i] * 0.125f);
        }
    }

    float mrow[8], lrow[8];
#pragma unroll
    for (int e = 0; e < 8; ++e) { mrow[e] = -1e30f; lrow[e] = 0.0f; }
    v8f o[4] = {};

    const int nb = (q0 + 47) / 32;      // 32-key blocks that intersect causal band

    for (int blkk = 0; blkk < nb; ++blkk) {
        const int k0 = blkk * 32;

        // ---- Stage V tile transposed: vT[d][key], bf16. lane -> key.
        {
            const float4* vp4 = reinterpret_cast<const float4*>(
                base + (size_t)(k0 + lane) * rstride + 2 * kD + h * kHD);
#pragma unroll
            for (int d4 = 0; d4 < 16; ++d4) {
                float4 f = vp4[d4];
                const int d = d4 * 4;
                vT[(d + 0) * 32 + lane] = tobf(f.x);
                vT[(d + 1) * 32 + lane] = tobf(f.y);
                vT[(d + 2) * 32 + lane] = tobf(f.z);
                vT[(d + 3) * 32 + lane] = tobf(f.w);
            }
        }

        // ---- S = Q @ K^T for two 16-key subtiles (keys k0.. and k0+16..)
        v8f s0 = {}, s1 = {};
        {
            const int dbase = halfsel * 16;
            const float* k0row = base + (size_t)(k0 + arow) * rstride + h * kHD + dbase;
            const float* k1row = base + (size_t)(k0 + 16 + arow) * rstride + h * kHD + dbase;
            const float4* k0p = reinterpret_cast<const float4*>(k0row);
            const float4* k1p = reinterpret_cast<const float4*>(k1row);
            v16bf kf;
            cvt4(kf, 0, k0p[0]); cvt4(kf, 4, k0p[1]); cvt4(kf, 8, k0p[2]); cvt4(kf, 12, k0p[3]);
            s0 = __builtin_amdgcn_wmma_f32_16x16x32_bf16(false, qa0, false, kf, (short)0, s0, false, false);
            cvt4(kf, 0, k0p[8]); cvt4(kf, 4, k0p[9]); cvt4(kf, 8, k0p[10]); cvt4(kf, 12, k0p[11]);
            s0 = __builtin_amdgcn_wmma_f32_16x16x32_bf16(false, qa1, false, kf, (short)0, s0, false, false);
            cvt4(kf, 0, k1p[0]); cvt4(kf, 4, k1p[1]); cvt4(kf, 8, k1p[2]); cvt4(kf, 12, k1p[3]);
            s1 = __builtin_amdgcn_wmma_f32_16x16x32_bf16(false, qa0, false, kf, (short)0, s1, false, false);
            cvt4(kf, 0, k1p[8]); cvt4(kf, 4, k1p[9]); cvt4(kf, 8, k1p[10]); cvt4(kf, 12, k1p[11]);
            s1 = __builtin_amdgcn_wmma_f32_16x16x32_bf16(false, qa1, false, kf, (short)0, s1, false, false);
        }

        // ---- Online softmax over the 32 keys of this block
#pragma unroll
        for (int e = 0; e < 8; ++e) {
            const int q = q0 + e + halfsel * 8;
            float a0 = ((k0 + arow)      <= q) ? s0[e] : -1e30f;
            float a1 = ((k0 + 16 + arow) <= q) ? s1[e] : -1e30f;
            float mx = fmaxf(a0, a1);
#pragma unroll
            for (int msk = 1; msk < 16; msk <<= 1)
                mx = fmaxf(mx, __shfl_xor(mx, msk, 32));
            const float mn   = fmaxf(mrow[e], mx);
            const float corr = __expf(mrow[e] - mn);
            const float p0 = __expf(a0 - mn);
            const float p1 = __expf(a1 - mn);
            float rs = p0 + p1;
#pragma unroll
            for (int msk = 1; msk < 16; msk <<= 1)
                rs += __shfl_xor(rs, msk, 32);
            lrow[e] = lrow[e] * corr + rs;
            mrow[e] = mn;
            o[0][e] *= corr; o[1][e] *= corr; o[2][e] *= corr; o[3][e] *= corr;
            const int m = e + halfsel * 8;
            pbuf[m * 32 + arow]      = tobf(p0);
            pbuf[m * 32 + 16 + arow] = tobf(p1);
        }
        __syncthreads();

        // ---- O += P @ V : P as A-fragment from LDS, V B-fragments from vT
        v16bf pa;
        {
            const int kb = halfsel * 16;
            const __bf16* pr = &pbuf[arow * 32];
#pragma unroll
            for (int i = 0; i < 8; ++i) {
                pa[i]     = pr[kb + i];
                pa[8 + i] = pr[kb + 8 + i];
            }
        }
        const int kbase = halfsel * 16;
#pragma unroll
        for (int t = 0; t < 4; ++t) {
            const __bf16* vp = &vT[(t * 16 + arow) * 32 + kbase];
            v16bf vb;
#pragma unroll
            for (int i = 0; i < 16; ++i) vb[i] = vp[i];
            o[t] = __builtin_amdgcn_wmma_f32_16x16x32_bf16(
                false, pa, false, vb, (short)0, o[t], false, false);
        }
        __syncthreads();   // vT/pbuf reused next iteration
    }

    // ---- Normalize and write out[b, q, h*64 + d]
    float inv[8];
#pragma unroll
    for (int e = 0; e < 8; ++e) inv[e] = 1.0f / lrow[e];
#pragma unroll
    for (int t = 0; t < 4; ++t) {
#pragma unroll
        for (int e = 0; e < 8; ++e) {
            const int row = q0 + e + halfsel * 8;
            const int col = h * kHD + t * 16 + arow;
            out[(size_t)(b * kT + row) * kD + col] = o[t][e] * inv[e];
        }
    }
}

// ---------------------------------------------------------------------------
// LayerNorm over rows of length D=1024. One 256-thread block per row.
// ---------------------------------------------------------------------------
__global__ __launch_bounds__(256) void layernorm_kernel(
    const float* __restrict__ x, const float* __restrict__ w,
    const float* __restrict__ b, float* __restrict__ y)
{
    __shared__ float red[256];
    const int row = blockIdx.x;
    const int tid = threadIdx.x;
    const float* xr = x + (size_t)row * kD;

    float v4[4];
    float s = 0.0f;
#pragma unroll
    for (int j = 0; j < 4; ++j) { v4[j] = xr[tid + j * 256]; s += v4[j]; }
    red[tid] = s;
    __syncthreads();
    for (int off = 128; off > 0; off >>= 1) {
        if (tid < off) red[tid] += red[tid + off];
        __syncthreads();
    }
    const float mean = red[0] * (1.0f / kD);
    __syncthreads();

    float sq = 0.0f;
#pragma unroll
    for (int j = 0; j < 4; ++j) { const float d = v4[j] - mean; sq += d * d; }
    red[tid] = sq;
    __syncthreads();
    for (int off = 128; off > 0; off >>= 1) {
        if (tid < off) red[tid] += red[tid + off];
        __syncthreads();
    }
    const float rstd = rsqrtf(red[0] * (1.0f / kD) + kEPS);

    float* yr = y + (size_t)row * kD;
#pragma unroll
    for (int j = 0; j < 4; ++j) {
        const int c = tid + j * 256;
        yr[c] = (v4[j] - mean) * rstd * w[c] + b[c];
    }
}

// ---------------------------------------------------------------------------
// Embedding: x[row] = tok_emb[inp[row]] + pos_emb[row % T]
// ---------------------------------------------------------------------------
__global__ __launch_bounds__(256) void embed_kernel(
    const int* __restrict__ inp, const float* __restrict__ tok,
    const float* __restrict__ pos, float* __restrict__ x)
{
    const int row = blockIdx.x;
    const int t = row % kT;
    const int token = inp[row];
    const float* te = tok + (size_t)token * kD;
    const float* pe = pos + (size_t)t * kD;
    float* xr = x + (size_t)row * kD;
#pragma unroll
    for (int j = 0; j < 4; ++j) {
        const int c = threadIdx.x + j * 256;
        xr[c] = te[c] + pe[c];
    }
}

// ---------------------------------------------------------------------------
// Per-row NLL: rowloss[row] = logsumexp(logits[row]) - logits[row][target]
// ---------------------------------------------------------------------------
__global__ __launch_bounds__(256) void loss_row_kernel(
    const float* __restrict__ logits, const int* __restrict__ targets,
    float* __restrict__ rowloss)
{
    __shared__ float red[256];
    const int row = blockIdx.x;
    const int tid = threadIdx.x;
    const float* lr = logits + (size_t)row * kV;

    float mx = -1e30f;
    for (int c = tid; c < kV; c += 256) mx = fmaxf(mx, lr[c]);
    red[tid] = mx;
    __syncthreads();
    for (int off = 128; off > 0; off >>= 1) {
        if (tid < off) red[tid] = fmaxf(red[tid], red[tid + off]);
        __syncthreads();
    }
    mx = red[0];
    __syncthreads();

    float s = 0.0f;
    for (int c = tid; c < kV; c += 256) s += __expf(lr[c] - mx);
    red[tid] = s;
    __syncthreads();
    for (int off = 128; off > 0; off >>= 1) {
        if (tid < off) red[tid] += red[tid + off];
        __syncthreads();
    }
    if (tid == 0) {
        const float lse = mx + __logf(red[0]);
        rowloss[row] = lse - lr[targets[row]];
    }
}

__global__ __launch_bounds__(256) void loss_reduce_kernel(
    const float* __restrict__ rowloss, float* __restrict__ loss)
{
    __shared__ float red[256];
    const int tid = threadIdx.x;
    float s = 0.0f;
    for (int i = tid; i < kM; i += 256) s += rowloss[i];
    red[tid] = s;
    __syncthreads();
    for (int off = 128; off > 0; off >>= 1) {
        if (tid < off) red[tid] += red[tid + off];
        __syncthreads();
    }
    if (tid == 0) *loss = red[0] * (1.0f / kM);
}

// ---------------------------------------------------------------------------
// Host: orchestrate the whole forward pass on `stream` (graph-capture safe).
// ---------------------------------------------------------------------------
extern "C" void kernel_launch(void* const* d_in, const int* in_sizes, int n_in,
                              void* d_out, int out_size, void* d_ws, size_t ws_size,
                              hipStream_t stream)
{
    (void)in_sizes; (void)n_in; (void)out_size; (void)ws_size;

    const int*   inp     = (const int*)d_in[0];
    const int*   targets = (const int*)d_in[1];
    const float* tok_emb = (const float*)d_in[2];
    const float* pos_emb = (const float*)d_in[3];
    const float* kqv_w   = (const float*)d_in[4];
    const float* kqv_b   = (const float*)d_in[5];
    const float* proj_w  = (const float*)d_in[6];
    const float* proj_b  = (const float*)d_in[7];
    const float* ff1_w   = (const float*)d_in[8];
    const float* ff1_b   = (const float*)d_in[9];
    const float* ff2_w   = (const float*)d_in[10];
    const float* ff2_b   = (const float*)d_in[11];
    const float* ln1_w   = (const float*)d_in[12];
    const float* ln1_b   = (const float*)d_in[13];
    const float* ln2_w   = (const float*)d_in[14];
    const float* ln2_b   = (const float*)d_in[15];
    const float* lnf_w   = (const float*)d_in[16];
    const float* lnf_b   = (const float*)d_in[17];

    float* logits = (float*)d_out;
    float* loss   = logits + (size_t)kM * kV;

    // Workspace carve-out (~160 MB)
    char* ws = (char*)d_ws;
    size_t off = 0;
    auto alloc = [&](size_t bytes) -> void* {
        void* p = ws + off;
        off += (bytes + 255) & ~(size_t)255;
        return p;
    };
    float* x       = (float*)alloc((size_t)kM * kD * 4);
    float* xn      = (float*)alloc((size_t)kM * kD * 4);
    float* kqv     = (float*)alloc((size_t)kM * 3 * kD * 4);
    float* attn    = (float*)alloc((size_t)kM * kD * 4);
    float* hff     = (float*)alloc((size_t)kM * 4 * kD * 4);
    float* rowloss = (float*)alloc((size_t)kM * 4);

    const dim3 blk256(256);

    embed_kernel<<<kM, blk256, 0, stream>>>(inp, tok_emb, pos_emb, x);

    for (int l = 0; l < kL; ++l) {
        layernorm_kernel<<<kM, blk256, 0, stream>>>(x, ln1_w + l * kD, ln1_b + l * kD, xn);
        gemm_bf16w<<<dim3(3 * kD / 128, kM / 128), blk256, 0, stream>>>(
            xn, kqv_w + (size_t)l * 3 * kD * kD, kqv_b + l * 3 * kD,
            nullptr, kqv, 3 * kD, kD, 0);
        attention_kernel<<<dim3(kT / 16, kB * kH), dim3(32), 0, stream>>>(kqv, attn);
        gemm_bf16w<<<dim3(kD / 128, kM / 128), blk256, 0, stream>>>(
            attn, proj_w + (size_t)l * kD * kD, proj_b + l * kD,
            x, x, kD, kD, 0);
        layernorm_kernel<<<kM, blk256, 0, stream>>>(x, ln2_w + l * kD, ln2_b + l * kD, xn);
        gemm_bf16w<<<dim3(4 * kD / 128, kM / 128), blk256, 0, stream>>>(
            xn, ff1_w + (size_t)l * 4 * kD * kD, ff1_b + l * 4 * kD,
            nullptr, hff, 4 * kD, kD, 1);
        gemm_bf16w<<<dim3(kD / 128, kM / 128), blk256, 0, stream>>>(
            hff, ff2_w + (size_t)l * kD * 4 * kD, ff2_b + l * kD,
            x, x, kD, 4 * kD, 0);
    }

    layernorm_kernel<<<kM, blk256, 0, stream>>>(x, lnf_w, lnf_b, xn);
    gemm_bf16w<<<dim3(kV / 128, kM / 128), blk256, 0, stream>>>(
        xn, tok_emb, nullptr, nullptr, logits, kV, kD, 0);

    loss_row_kernel<<<kM, blk256, 0, stream>>>(logits, targets, rowloss);
    loss_reduce_kernel<<<1, blk256, 0, stream>>>(rowloss, loss);
}
